// LFGNN_88098369176053
// MI455X (gfx1250) — compile-verified, matrix operand order checked by reading, not run on hardware
//
#include <hip/hip_runtime.h>

// ---------------------------------------------------------------------------
// CDNA5 / gfx1250 GraphSAGE autoencoder.
//  - Dense GEMMs: v_wmma_f32_16x16x32_bf16, f32 accumulate.
//  - Weights pre-packed into per-lane WMMA B-fragment order -> each lane loads
//    its whole fragment as one 32B vector load (2x global_load_b128).
//  - One wave computes a 16x64 output strip (4 independent accumulator chains
//    per A fragment, covering the bf16-WMMA RAW hazard distance).
//  - Graph aggregation (mean segment_sum) stays f32 with global atomics;
//    inv_deg folded into the scatter (linearity of the sum).
//  - RELU / full-column predicates are compile-time -> branch-free epilogue.
// ---------------------------------------------------------------------------

typedef __attribute__((ext_vector_type(16))) __bf16 v16bf;
typedef __attribute__((ext_vector_type(8)))  float  v8f;

// -------------------------- small utility kernels --------------------------

__global__ void zero_f32(float* __restrict__ p, long n) {
  long i = (long)blockIdx.x * blockDim.x + threadIdx.x;
  if (i < n) p[i] = 0.0f;
}

// Repack f32 weight [K][scols] into bf16 WMMA B-fragment order:
//   dst[ ((tile*(K/32) + ks)*32 + lane)*16 + j ]
// where col = tile*16 + lane%16, kb = ks*32 + (lane/16)*8,
//       k = kb + j            (j in 0..7)
//       k = kb + 16 + (j-8)   (j in 8..15)
// Columns >= scols are zero-padded (used for W_dec2: 8 -> 16 cols).
__global__ void pack_weights(const float* __restrict__ src, __bf16* __restrict__ dst,
                             int K, int scols, int colTiles) {
  const int ksteps = K >> 5;
  int idx = blockIdx.x * blockDim.x + threadIdx.x;
  if (idx >= colTiles * ksteps * 32 * 16) return;
  int j    = idx & 15;
  int lane = (idx >> 4) & 31;
  int rem  = idx >> 9;
  int ks   = rem % ksteps;
  int ct   = rem / ksteps;
  int col  = (ct << 4) + (lane & 15);
  int kb   = (ks << 5) + ((lane >> 4) << 3);
  int k    = kb + j + ((j < 8) ? 0 : 8);
  float v  = (col < scols) ? src[(size_t)k * scols + col] : 0.0f;
  dst[idx] = (__bf16)v;
}

__global__ void deg_accum(const int* __restrict__ ei, float* __restrict__ deg, int E) {
  int e = blockIdx.x * blockDim.x + threadIdx.x;
  if (e < E) atomicAdd(&deg[ei[E + e]], 1.0f);   // ei[1][e] = dst
}

__global__ void deg_invert(float* __restrict__ deg, int n) {
  int i = blockIdx.x * blockDim.x + threadIdx.x;
  if (i < n) { float d = deg[i]; deg[i] = (d > 0.0f) ? (1.0f / d) : 0.0f; }
}

// One wave per edge; lane l moves h[src][4l..4l+3] * inv_deg[dst] into
// agg[dst] with f32 global atomics (mean-aggregation, scale folded in).
__global__ void scatter_accum(const int* __restrict__ ei, const float* __restrict__ h,
                              const float* __restrict__ invdeg, float* __restrict__ agg,
                              int E) {
  int t    = blockIdx.x * blockDim.x + threadIdx.x;
  int edge = t >> 5;
  int lane = t & 31;
  if (edge >= E) return;
  int src = ei[edge];
  int dst = ei[E + edge];
  float s  = invdeg[dst];
  float4 v = ((const float4*)(h + (size_t)src * 128))[lane];
  float* a = agg + (size_t)dst * 128 + lane * 4;
  atomicAdd(a + 0, v.x * s);
  atomicAdd(a + 1, v.y * s);
  atomicAdd(a + 2, v.z * s);
  atomicAdd(a + 3, v.w * s);
}

// -------------------------- WMMA GEMM --------------------------------------

// A fragment (16-bit A, 16x32): lane%16 = row; lane/16 selects K-half:
//   elems 0..7 = K = kh+0..7, elems 8..15 = K = kh+16..23 (kh = 0 or 8).
// ap points at A[row] + ks*32 + kh; two contiguous 32B runs -> b128 loads,
// converted with packed v_cvt_pk_bf16_f32.
__device__ __forceinline__ v16bf load_a_frag(const float* __restrict__ ap) {
  v16bf a;
#pragma unroll
  for (int i = 0; i < 8; ++i) {
    a[i]     = (__bf16)ap[i];
    a[i + 8] = (__bf16)ap[16 + i];
  }
  return a;
}

// out = act( A1 @ W1 + A2 @ W2 + bias ).  One wave computes 16 rows x CT*16
// cols; 8 waves / 256-thread block.  K1/K2 compile-time -> fully unrolled
// straight-line v_wmma; RELU/FULL compile-time -> branch-free epilogue.
// All control flow before the WMMAs is wave-uniform (EXEC all-ones).
template <int K1, int K2, int CT, bool RELU, bool FULL>
__global__ void __launch_bounds__(256)
wmma_gemm(const float* __restrict__ A1, const __bf16* __restrict__ W1,
          const float* __restrict__ A2, const __bf16* __restrict__ W2,
          const float* __restrict__ bias, float* __restrict__ out,
          int nrows, int out_cols, int out_ld) {
  const int colTiles  = (out_cols + 15) >> 4;
  const int colGroups = colTiles / CT;
  const int rowTiles  = nrows >> 4;              // nrows multiple of 16
  const int wave = threadIdx.x >> 5;
  const int lane = threadIdx.x & 31;
  const int work = blockIdx.x * 8 + wave;
  if (work >= rowTiles * colGroups) return;      // wave-uniform exit
  const int rowTile = work / colGroups;
  const int cg      = work - rowTile * colGroups;
  const int row     = (rowTile << 4) + (lane & 15);
  const int kHalf   = (lane >> 4) << 3;          // 0 or 8

  v8f acc[CT];
#pragma unroll
  for (int ct = 0; ct < CT; ++ct) acc[ct] = {};

  {
    const float* arow = A1 + (size_t)row * K1 + kHalf;
#pragma unroll
    for (int ks = 0; ks < K1 / 32; ++ks) {
      v16bf a = load_a_frag(arow + ks * 32);
#pragma unroll
      for (int ct = 0; ct < CT; ++ct) {
        const v16bf b = *(const v16bf*)(
            W1 + ((((size_t)(cg * CT + ct) * (K1 / 32)) + ks) * 32 + lane) * 16);
        acc[ct] = __builtin_amdgcn_wmma_f32_16x16x32_bf16(
            false, a, false, b, (short)0, acc[ct], false, false);
      }
    }
  }
  if constexpr (K2 > 0) {
    const float* arow = A2 + (size_t)row * K2 + kHalf;
#pragma unroll
    for (int ks = 0; ks < K2 / 32; ++ks) {
      v16bf a = load_a_frag(arow + ks * 32);
#pragma unroll
      for (int ct = 0; ct < CT; ++ct) {
        const v16bf b = *(const v16bf*)(
            W2 + ((((size_t)(cg * CT + ct) * (K2 / 32)) + ks) * 32 + lane) * 16);
        acc[ct] = __builtin_amdgcn_wmma_f32_16x16x32_bf16(
            false, a, false, b, (short)0, acc[ct], false, false);
      }
    }
  }

  // C/D layout: lanes 0-15 -> M = r, lanes 16-31 -> M = r + 8 (VGPR r).
  const int mb = (lane >> 4) << 3;
#pragma unroll
  for (int ct = 0; ct < CT; ++ct) {
    const int  col = ((cg * CT + ct) << 4) + (lane & 15);
    const bool ok  = FULL || (col < out_cols);
    const float bv = ok ? bias[col] : 0.0f;
#pragma unroll
    for (int r = 0; r < 8; ++r) {
      float v = acc[ct][r] + bv;
      if constexpr (RELU) v = fmaxf(v, 0.0f);
      if (ok) out[(size_t)((rowTile << 4) + mb + r) * out_ld + col] = v;
    }
  }
}

// -------------------------- host-side launch -------------------------------

extern "C" void kernel_launch(void* const* d_in, const int* in_sizes, int n_in,
                              void* d_out, int out_size, void* d_ws, size_t ws_size,
                              hipStream_t stream) {
  constexpr int N = 100000, E = 1600000, H = 128;

  const float* x      = (const float*)d_in[0];
  const int*   ei     = (const int*)  d_in[1];
  const float* W_enc1 = (const float*)d_in[2];
  const float* b_enc1 = (const float*)d_in[3];
  const float* W_enc2 = (const float*)d_in[4];
  const float* b_enc2 = (const float*)d_in[5];
  const float* W_l    = (const float*)d_in[6];
  const float* b_l    = (const float*)d_in[7];
  const float* W_r    = (const float*)d_in[8];
  const float* W_dec1 = (const float*)d_in[9];
  const float* b_dec1 = (const float*)d_in[10];
  const float* W_dec2 = (const float*)d_in[11];
  const float* b_dec2 = (const float*)d_in[12];

  float* out_dec = (float*)d_out;                       // [N, 8]
  float* h_final = (float*)d_out + (size_t)N * 8;       // [N, 128]

  // workspace carve-up (all blocks 32B-aligned: sizes are multiples of 32B)
  char* w = (char*)d_ws;
  float* h0     = (float*)w; w += (size_t)N * H * sizeof(float);
  float* h1     = (float*)w; w += (size_t)N * H * sizeof(float);
  float* agg    = (float*)w; w += (size_t)N * H * sizeof(float);
  float* invdeg = (float*)w; w += (size_t)N * sizeof(float);
  __bf16* wb      = (__bf16*)w;
  __bf16* wb_enc1 = wb; wb += 32 * 128;       // 8 tiles * 1 kstep * 512
  __bf16* wb_enc2 = wb; wb += 128 * 128;      // 8 tiles * 4 ksteps * 512
  __bf16* wb_l    = wb; wb += 3 * 128 * 128;
  __bf16* wb_r    = wb; wb += 3 * 128 * 128;
  __bf16* wb_dec1 = wb; wb += 128 * 128;
  __bf16* wb_dec2 = wb; wb += 128 * 16;       // 1 tile * 4 ksteps * 512

  auto cg = [](long total) { return (unsigned)((total + 255) / 256); };
  auto gg = [](int colGroups) {                // GEMM grid: 8 waves per block
    return (unsigned)(((long)(N / 16) * colGroups + 7) / 8);
  };

  // 1) weights -> packed bf16 B-fragments
  pack_weights<<<cg(32 * 128),  256, 0, stream>>>(W_enc1, wb_enc1, 32,  128, 8);
  pack_weights<<<cg(128 * 128), 256, 0, stream>>>(W_enc2, wb_enc2, 128, 128, 8);
  for (int l = 0; l < 3; ++l) {
    pack_weights<<<cg(128 * 128), 256, 0, stream>>>(
        W_l + (size_t)l * 128 * 128, wb_l + (size_t)l * 128 * 128, 128, 128, 8);
    pack_weights<<<cg(128 * 128), 256, 0, stream>>>(
        W_r + (size_t)l * 128 * 128, wb_r + (size_t)l * 128 * 128, 128, 128, 8);
  }
  pack_weights<<<cg(128 * 128), 256, 0, stream>>>(W_dec1, wb_dec1, 128, 128, 8);
  pack_weights<<<cg(128 * 16),  256, 0, stream>>>(W_dec2, wb_dec2, 128, 8,   1);

  // 2) in-degree -> inv_deg
  zero_f32<<<cg(N),   256, 0, stream>>>(invdeg, N);
  deg_accum<<<cg(E),  256, 0, stream>>>(ei, invdeg, E);
  deg_invert<<<cg(N), 256, 0, stream>>>(invdeg, N);

  // 3) encoder
  wmma_gemm<32, 0, 4, true, true><<<gg(2), 256, 0, stream>>>(
      x, wb_enc1, nullptr, nullptr, b_enc1, h0, N, 128, 128);
  wmma_gemm<128, 0, 4, true, true><<<gg(2), 256, 0, stream>>>(
      h0, wb_enc2, nullptr, nullptr, b_enc2, h1, N, 128, 128);

  // 4) SAGE layers: h = relu(agg@W_l + b_l + h@W_r), dual-K fused GEMM
  float* hcur  = h1;
  float* hnext = h0;
  for (int l = 0; l < 3; ++l) {
    zero_f32<<<cg((long)N * H), 256, 0, stream>>>(agg, (long)N * H);
    scatter_accum<<<cg((long)E * 32), 256, 0, stream>>>(ei, hcur, invdeg, agg, E);
    float* dst = (l == 2) ? h_final : hnext;
    wmma_gemm<128, 128, 4, true, true><<<gg(2), 256, 0, stream>>>(
        agg, wb_l + (size_t)l * 128 * 128,
        hcur, wb_r + (size_t)l * 128 * 128,
        b_l + l * 128, dst, N, 128, 128);
    float* tmp = hcur; hcur = dst; hnext = tmp;
  }

  // 5) decoder
  wmma_gemm<128, 0, 4, true, true><<<gg(2), 256, 0, stream>>>(
      h_final, wb_dec1, nullptr, nullptr, b_dec1, h0, N, 128, 128);
  wmma_gemm<128, 0, 1, false, false><<<gg(1), 256, 0, stream>>>(
      h0, wb_dec2, nullptr, nullptr, b_dec2, out_dec, N, 8, 8);
}
